// ARC_65249143160995
// MI455X (gfx1250) — matrix operand order repaired and checked by reading, not compiled, not run on hardware
//
#include <hip/hip_runtime.h>
#include <hip/hip_bf16.h>

// ---------------------------------------------------------------------------
// Fused graph-network for MI455X (gfx1250, wave32, WMMA).
// All MLP layers run as V_WMMA_F32_16X16X32_F16 (f32 accumulate). Weights are
// pre-swizzled per wave into WMMA B-fragments held in VGPRs. A 1KB per-wave
// LDS tile (16 rows x 32 halfs) is used to convert D-fragment layout into the
// next layer's A-fragment layout (cross-lane relayout via ds ops).
// Segment sums use global f32 atomics (edge->node) and an LDS per-block
// pre-reduction (node->graph, only 64*19 destinations).
// SELU is branchless: scale*max(x,0) + scale*alpha*(expm1(min(x,0))).
// ---------------------------------------------------------------------------

typedef __attribute__((ext_vector_type(16))) _Float16 v16h;
typedef __attribute__((ext_vector_type(8)))  _Float16 v8h;
typedef __attribute__((ext_vector_type(8)))  float    v8f;

#define NGRAPH 64
#define GCOLS  19

__device__ __forceinline__ int laneid() { return (int)(threadIdx.x & 31u); }

__device__ __forceinline__ v8f wmma16x16x32(v16h a, v16h b, v8f c) {
  // D = A(16x32 f16) x B(32x16 f16) + C(16x16 f32)
  return __builtin_amdgcn_wmma_f32_16x16x32_f16(false, a, false, b,
                                                (short)0, c, false, false);
}

// Branchless SELU: exact since exp(0)-1 == 0.
__device__ __forceinline__ float selu_f(float x) {
  const float kScale = 1.0507009873554805f;
  const float kSA    = 1.7580993408473766f;   // scale * alpha
  const float p = fmaxf(x, 0.0f);
  const float m = fminf(x, 0.0f);
  return fmaf(kScale, p, kSA * (__expf(m) - 1.0f));
}

// Build a B fragment (K=32 x N=16 tile) from row-major W[Fin][Fout].
// Layout (16-bit B, 32x16): lane l<16 holds column n=l, K 0..7 & 16..23;
// lane l>=16 holds column n=l-16, K 8..15 & 24..31. Zero-padded beyond Fin/Fout.
__device__ __forceinline__ v16h make_b_frag(const float* W, int Fin, int Fout,
                                            int ntile) {
  const int l  = laneid();
  const int n  = (l & 15) + ntile * 16;
  const int k0 = (l < 16) ? 0 : 8;
  v16h b;
#pragma unroll
  for (int i = 0; i < 16; ++i) {
    const int k = k0 + ((i < 8) ? i : 8 + i);
    const float w = (k < Fin && n < Fout) ? W[k * Fout + n] : 0.0f;
    b[i] = (_Float16)w;
  }
  return b;
}

__device__ __forceinline__ float make_bias(const float* bias, int Fout, int ntile) {
  const int n = (laneid() & 15) + ntile * 16;
  return (n < Fout) ? bias[n] : 0.0f;
}

// Zero a 16x32-half staging tile (1KB): 2 x b128 per lane.
__device__ __forceinline__ void lds_zero_tile(_Float16* buf) {
  v8h z = {};
  ((v8h*)buf)[laneid()]      = z;
  ((v8h*)buf)[laneid() + 32] = z;
}

// Load A fragment (16x32 f16) from row-major staging tile (row stride 32 halfs).
// ISA A layout: lane l<16 -> row l, K 0..7 & 16..23; lane l>=16 -> row l-16,
// K 8..15 & 24..31. Two ds_load_b128 per call.
__device__ __forceinline__ v16h lds_load_a(const _Float16* buf) {
  const int l = laneid();
  const _Float16* p = buf + (l & 15) * 32 + ((l < 16) ? 0 : 8);
  v8h lo = *(const v8h*)(p);
  v8h hi = *(const v8h*)(p + 16);
  v16h a;
#pragma unroll
  for (int i = 0; i < 8; ++i) { a[i] = lo[i]; a[i + 8] = hi[i]; }
  return a;
}

// Store D fragment (+bias, optional SELU) as f16 into staging tile at
// [row][col_off + n]; lanes with n >= ncols store zero (N padding).
// D layout: lane l, VGPR r -> (m = r + 8*(l/16), n = l%16).
__device__ __forceinline__ void lds_store_d(_Float16* buf, v8f d, float bias,
                                            bool do_selu, int ncols, int col_off) {
  const int l  = laneid();
  const int n  = l & 15;
  const int mo = (l < 16) ? 0 : 8;
#pragma unroll
  for (int r = 0; r < 8; ++r) {
    float x = d[r] + bias;
    if (do_selu) x = selu_f(x);
    buf[(r + mo) * 32 + col_off + n] = (n < ncols) ? (_Float16)x : (_Float16)0.0f;
  }
}

// ---------------------------------------------------------------------------
// Kernel 0: zero the workspace accumulators.
// ---------------------------------------------------------------------------
__global__ void k_zero(float* p, size_t cnt) {
  size_t i = (size_t)blockIdx.x * blockDim.x + threadIdx.x;
  const size_t stride = (size_t)gridDim.x * blockDim.x;
  for (; i < cnt; i += stride) p[i] = 0.0f;
}

// ---------------------------------------------------------------------------
// Kernel 1: edge phase. gather -> pn MLP -> 3x ue MLP -> atomic scatter-add
// into per-receiver accumulator [N,3].
// ---------------------------------------------------------------------------
struct EdgeArgs {
  const float* nodes; const float* edges;
  const int* senders; const int* receivers;
  const float *pnW0,*pnB0,*pnW1,*pnB1,*pnW2,*pnB2;
  const float *ueW0,*ueB0,*ueW1,*ueB1,*ueW2,*ueB2;
  float* node_acc;
  int E;
};

__launch_bounds__(256)
__global__ void k_edges(EdgeArgs A) {
  __shared__ __align__(16) _Float16 smem[8][2][16 * 32];
  const int wid = threadIdx.x >> 5;
  _Float16* bufA = smem[wid][0];
  _Float16* bufB = smem[wid][1];
  const int l  = laneid();
  const int n  = l & 15;
  const int mo = (l < 16) ? 0 : 8;

  // Pre-swizzled weight fragments (live in VGPRs for the whole kernel).
  v16h Bpn0  = make_b_frag(A.pnW0,  6, 16, 0); float bpn0  = make_bias(A.pnB0, 16, 0);
  v16h Bpn1a = make_b_frag(A.pnW1, 16, 32, 0); float bpn1a = make_bias(A.pnB1, 32, 0);
  v16h Bpn1b = make_b_frag(A.pnW1, 16, 32, 1); float bpn1b = make_bias(A.pnB1, 32, 1);
  v16h Bpn2  = make_b_frag(A.pnW2, 32, 16, 0); float bpn2  = make_bias(A.pnB2, 16, 0);
  v16h Bue0a = make_b_frag(A.ueW0, 19, 32, 0); float bue0a = make_bias(A.ueB0, 32, 0);
  v16h Bue0b = make_b_frag(A.ueW0, 19, 32, 1); float bue0b = make_bias(A.ueB0, 32, 1);
  v16h Bue1  = make_b_frag(A.ueW1, 32, 16, 0); float bue1  = make_bias(A.ueB1, 16, 0);
  v16h Bue2  = make_b_frag(A.ueW2, 16,  3, 0); float bue2  = make_bias(A.ueB2,  3, 0);

  const int ntiles = (A.E + 15) >> 4;
  const int gw = blockIdx.x * (blockDim.x >> 5) + wid;
  const int nw = gridDim.x * (blockDim.x >> 5);
  const v8f c0 = {};

  for (int t = gw; t < ntiles; t += nw) {
    const int e0 = t << 4;

    // Stage collected = [nodes[rcv], nodes[snd]] into cols 0..5 (random gather).
    // Zero-fill guards against NaN bit patterns in untouched K-pad columns;
    // all later intermediates are finite so B-fragment zero-padding handles
    // the rest (garbage * 0 == 0 for finite garbage).
    lds_zero_tile(bufA);
    if (l < 16) {
      int e = e0 + l; if (e >= A.E) e = A.E - 1;
      const int rv = A.receivers[e], sv = A.senders[e];
      _Float16* row = bufA + l * 32;
      row[0] = (_Float16)A.nodes[rv * 3 + 0];
      row[1] = (_Float16)A.nodes[rv * 3 + 1];
      row[2] = (_Float16)A.nodes[rv * 3 + 2];
      row[3] = (_Float16)A.nodes[sv * 3 + 0];
      row[4] = (_Float16)A.nodes[sv * 3 + 1];
      row[5] = (_Float16)A.nodes[sv * 3 + 2];
    }

    // pn MLP: 6 -> 16 -> 32 -> 16 (SELU between, none at end)
    v16h a = lds_load_a(bufA);
    v8f d  = wmma16x16x32(a, Bpn0, c0);
    lds_store_d(bufA, d, bpn0, true, 16, 0);
    a = lds_load_a(bufA);
    v8f d0 = wmma16x16x32(a, Bpn1a, c0);
    v8f d1 = wmma16x16x32(a, Bpn1b, c0);
    lds_store_d(bufA, d0, bpn1a, true, 16, 0);
    lds_store_d(bufA, d1, bpn1b, true, 16, 16);
    a = lds_load_a(bufA);
    v8f h1n = wmma16x16x32(a, Bpn2, c0);
#pragma unroll
    for (int r = 0; r < 8; ++r) h1n[r] += bpn2;     // last layer: bias, no SELU

    // h_edges initial value from global edges [E,3]
    v8f he;
#pragma unroll
    for (int r = 0; r < 8; ++r) {
      int e = e0 + r + mo; if (e >= A.E) e = A.E - 1;
      he[r] = (n < 3) ? A.edges[e * 3 + n] : 0.0f;
    }

    // 3x update_edges: concat([he(3), h1n(16)]) -> 19 -> 32 -> 16 -> 3
    for (int it = 0; it < 3; ++it) {
      lds_store_d(bufA, he,  0.0f, false, 3, 0);    // cols 0..2 (pad zeros 3..15)
      lds_store_d(bufA, h1n, 0.0f, false, 16, 3);   // cols 3..18
      a  = lds_load_a(bufA);
      d0 = wmma16x16x32(a, Bue0a, c0);
      d1 = wmma16x16x32(a, Bue0b, c0);
      lds_store_d(bufB, d0, bue0a, true, 16, 0);
      lds_store_d(bufB, d1, bue0b, true, 16, 16);
      a = lds_load_a(bufB);
      d = wmma16x16x32(a, Bue1, c0);
      lds_store_d(bufB, d, bue1, true, 16, 0);      // cols16..31 stale-but-finite,
      a = lds_load_a(bufB);                         // Bue2 zero-pads k>=16
      d = wmma16x16x32(a, Bue2, c0);
#pragma unroll
      for (int r = 0; r < 8; ++r) he[r] = d[r] + bue2;  // no SELU at end
    }

    // EdgesToNodes segment-sum: scatter-add into receiver accumulator.
    if (n < 3) {
#pragma unroll
      for (int r = 0; r < 8; ++r) {
        const int e = e0 + r + mo;
        if (e < A.E) atomicAdd(&A.node_acc[A.receivers[e] * 3 + n], he[r]);
      }
    }
  }
}

// ---------------------------------------------------------------------------
// Kernel 2: node phase. pe MLP -> 3x un MLP -> per-graph reduction (LDS
// pre-reduction, then one global atomic sweep per block).
// ---------------------------------------------------------------------------
struct NodeArgs {
  const float* nodes; const int* gids;
  const float *peW0,*peB0,*peW1,*peB1,*peW2,*peB2;
  const float *unW0,*unB0,*unW1,*unB1,*unW2,*unB2;
  const float* node_acc;
  float* glob_acc;
  int N;
};

__launch_bounds__(256)
__global__ void k_nodes(NodeArgs A) {
  __shared__ __align__(16) _Float16 smem[8][2][16 * 32];
  __shared__ float gacc[NGRAPH * GCOLS];
  const int wid = threadIdx.x >> 5;
  _Float16* bufA = smem[wid][0];
  _Float16* bufB = smem[wid][1];
  const int l  = laneid();
  const int n  = l & 15;
  const int mo = (l < 16) ? 0 : 8;

  for (int i = threadIdx.x; i < NGRAPH * GCOLS; i += blockDim.x) gacc[i] = 0.0f;
  __syncthreads();

  v16h Bpe0  = make_b_frag(A.peW0,  3, 16, 0); float bpe0  = make_bias(A.peB0, 16, 0);
  v16h Bpe1a = make_b_frag(A.peW1, 16, 32, 0); float bpe1a = make_bias(A.peB1, 32, 0);
  v16h Bpe1b = make_b_frag(A.peW1, 16, 32, 1); float bpe1b = make_bias(A.peB1, 32, 1);
  v16h Bpe2  = make_b_frag(A.peW2, 32, 16, 0); float bpe2  = make_bias(A.peB2, 16, 0);
  v16h Bun0a = make_b_frag(A.unW0, 19, 32, 0); float bun0a = make_bias(A.unB0, 32, 0);
  v16h Bun0b = make_b_frag(A.unW0, 19, 32, 1); float bun0b = make_bias(A.unB0, 32, 1);
  v16h Bun1  = make_b_frag(A.unW1, 32, 16, 0); float bun1  = make_bias(A.unB1, 16, 0);
  v16h Bun2  = make_b_frag(A.unW2, 16,  3, 0); float bun2  = make_bias(A.unB2,  3, 0);

  const int ntiles = (A.N + 15) >> 4;
  const int gw = blockIdx.x * (blockDim.x >> 5) + wid;
  const int nw = gridDim.x * (blockDim.x >> 5);
  const v8f c0 = {};

  for (int t = gw; t < ntiles; t += nw) {
    const int n0 = t << 4;

    // Stage edges_of_nodes [16,3] into cols 0..2.
    lds_zero_tile(bufA);
    if (l < 16) {
      int idx = n0 + l; if (idx >= A.N) idx = A.N - 1;
      _Float16* row = bufA + l * 32;
      row[0] = (_Float16)A.node_acc[idx * 3 + 0];
      row[1] = (_Float16)A.node_acc[idx * 3 + 1];
      row[2] = (_Float16)A.node_acc[idx * 3 + 2];
    }

    // pe MLP: 3 -> 16 -> 32 -> 16
    v16h a = lds_load_a(bufA);
    v8f d  = wmma16x16x32(a, Bpe0, c0);
    lds_store_d(bufA, d, bpe0, true, 16, 0);
    a = lds_load_a(bufA);
    v8f d0 = wmma16x16x32(a, Bpe1a, c0);
    v8f d1 = wmma16x16x32(a, Bpe1b, c0);
    lds_store_d(bufA, d0, bpe1a, true, 16, 0);
    lds_store_d(bufA, d1, bpe1b, true, 16, 16);
    a = lds_load_a(bufA);
    v8f h2e = wmma16x16x32(a, Bpe2, c0);
#pragma unroll
    for (int r = 0; r < 8; ++r) h2e[r] += bpe2;

    // h2_nodes initial value from global nodes [N,3]
    v8f h2n;
#pragma unroll
    for (int r = 0; r < 8; ++r) {
      int idx = n0 + r + mo; if (idx >= A.N) idx = A.N - 1;
      h2n[r] = (n < 3) ? A.nodes[idx * 3 + n] : 0.0f;
    }

    // 3x update_nodes: concat([h2n(3), h2e(16)]) -> 19 -> 32 -> 16 -> 3
    for (int it = 0; it < 3; ++it) {
      lds_store_d(bufA, h2n, 0.0f, false, 3, 0);
      lds_store_d(bufA, h2e, 0.0f, false, 16, 3);
      a  = lds_load_a(bufA);
      d0 = wmma16x16x32(a, Bun0a, c0);
      d1 = wmma16x16x32(a, Bun0b, c0);
      lds_store_d(bufB, d0, bun0a, true, 16, 0);
      lds_store_d(bufB, d1, bun0b, true, 16, 16);
      a = lds_load_a(bufB);
      d = wmma16x16x32(a, Bun1, c0);
      lds_store_d(bufB, d, bun1, true, 16, 0);
      a = lds_load_a(bufB);
      d = wmma16x16x32(a, Bun2, c0);
#pragma unroll
      for (int r = 0; r < 8; ++r) h2n[r] = d[r] + bun2;
    }

    // per_node = concat([h2n(3), h2e(16)]) -> LDS per-graph pre-reduction.
#pragma unroll
    for (int r = 0; r < 8; ++r) {
      const int idx = n0 + r + mo;
      if (idx < A.N) {
        const int g = A.gids[idx];
        atomicAdd(&gacc[g * GCOLS + 3 + n], h2e[r]);          // cols 3..18
        if (n < 3) atomicAdd(&gacc[g * GCOLS + n], h2n[r]);   // cols 0..2
      }
    }
  }

  __syncthreads();
  for (int i = threadIdx.x; i < NGRAPH * GCOLS; i += blockDim.x)
    atomicAdd(&A.glob_acc[i], gacc[i]);
}

// ---------------------------------------------------------------------------
// Kernel 3: predict MLP on [64,19] globals -> [64,9]. One wave, 4 row-tiles.
// ---------------------------------------------------------------------------
struct PredArgs {
  const float* glob_acc;
  const float *prW0,*prB0,*prW1,*prB1,*prW2,*prB2;
  float* out;
};

__launch_bounds__(32)
__global__ void k_predict(PredArgs A) {
  __shared__ __align__(16) _Float16 buf[16 * 32];
  const int l  = laneid();
  const int n  = l & 15;
  const int mo = (l < 16) ? 0 : 8;

  v16h B0 = make_b_frag(A.prW0, 19, 10, 0); float b0 = make_bias(A.prB0, 10, 0);
  v16h B1 = make_b_frag(A.prW1, 10, 10, 0); float b1 = make_bias(A.prB1, 10, 0);
  v16h B2 = make_b_frag(A.prW2, 10,  9, 0); float b2 = make_bias(A.prB2,  9, 0);
  const v8f c0 = {};

  for (int t = 0; t < 4; ++t) {
    lds_zero_tile(buf);
    if (l < 16) {
      const float* g = A.glob_acc + (t * 16 + l) * GCOLS;
      _Float16* row = buf + l * 32;
#pragma unroll
      for (int c = 0; c < GCOLS; ++c) row[c] = (_Float16)g[c];
    }
    v16h a = lds_load_a(buf);
    v8f d  = wmma16x16x32(a, B0, c0);
    lds_store_d(buf, d, b0, true, 10, 0);
    a = lds_load_a(buf);
    d = wmma16x16x32(a, B1, c0);
    lds_store_d(buf, d, b1, true, 10, 0);
    a = lds_load_a(buf);
    d = wmma16x16x32(a, B2, c0);
    if (n < 9) {
#pragma unroll
      for (int r = 0; r < 8; ++r)
        A.out[(t * 16 + r + mo) * 9 + n] = d[r] + b2;   // last layer: no SELU
    }
  }
}

// ---------------------------------------------------------------------------
// Launch. Input order = setup_inputs() dict order:
//   0 nodes[N,3] f32, 1 edges[E,3] f32, 2 senders[E] i32, 3 receivers[E] i32,
//   4 node_gids[N] i32, then params flattened pn,ue,pe,un,pr x (W,b) x 3.
// ---------------------------------------------------------------------------
extern "C" void kernel_launch(void* const* d_in, const int* in_sizes, int n_in,
                              void* d_out, int out_size, void* d_ws, size_t ws_size,
                              hipStream_t stream) {
  const float* nodes     = (const float*)d_in[0];
  const float* edges     = (const float*)d_in[1];
  const int*   senders   = (const int*)d_in[2];
  const int*   receivers = (const int*)d_in[3];
  const int*   gids      = (const int*)d_in[4];
  const int E = in_sizes[2];
  const int N = in_sizes[4];

  const float* P[30];
  for (int i = 0; i < 30; ++i) P[i] = (const float*)d_in[5 + i];
  const float* const* pn = P + 0;
  const float* const* ue = P + 6;
  const float* const* pe = P + 12;
  const float* const* un = P + 18;
  const float* const* pr = P + 24;

  float* node_acc = (float*)d_ws;                    // [N,3]
  float* glob_acc = node_acc + (size_t)N * 3;        // [64,19]

  k_zero<<<512, 256, 0, stream>>>(node_acc, (size_t)N * 3 + NGRAPH * GCOLS);

  EdgeArgs ea = { nodes, edges, senders, receivers,
                  pn[0], pn[1], pn[2], pn[3], pn[4], pn[5],
                  ue[0], ue[1], ue[2], ue[3], ue[4], ue[5],
                  node_acc, E };
  k_edges<<<1024, 256, 0, stream>>>(ea);

  NodeArgs na = { nodes, gids,
                  pe[0], pe[1], pe[2], pe[3], pe[4], pe[5],
                  un[0], un[1], un[2], un[3], un[4], un[5],
                  node_acc, glob_acc, N };
  k_nodes<<<256, 256, 0, stream>>>(na);

  PredArgs pa = { glob_acc, pr[0], pr[1], pr[2], pr[3], pr[4], pr[5],
                  (float*)d_out };
  k_predict<<<1, 32, 0, stream>>>(pa);
}
